// HybridModel_90220083019996
// MI455X (gfx1250) — matrix operand order repaired and checked by reading, not compiled, not run on hardware
//
#include <hip/hip_runtime.h>

// ---------------------------------------------------------------------------
// Model dims (match reference)
// ---------------------------------------------------------------------------
#define DIMC   256
#define SEQC   64
#define BATCHC 64
#define TOK    4096      // BATCH*SEQ
#define HEADS  4
#define HD     64
#define FFC    1024
#define VOC    32000
#define NL     12

typedef __attribute__((ext_vector_type(16))) _Float16 v16h;
typedef __attribute__((ext_vector_type(8)))  _Float16 v8h;
typedef __attribute__((ext_vector_type(8)))  float    v8f;

// A fragment 16x32 f16: lane m%16 holds row m; lanes<16 hold K {0..7,16..23},
// lanes>=16 hold K {8..15,24..31}  (CDNA5 ISA 7.12.2)
__device__ __forceinline__ v16h load_a_frag(const _Float16* __restrict__ arow,
                                            int k, int hf) {
  v8h lo = *(const v8h*)(arow + k + 8 * hf);
  v8h hi = *(const v8h*)(arow + k + 16 + 8 * hf);
  return __builtin_shufflevector(lo, hi, 0,1,2,3,4,5,6,7,8,9,10,11,12,13,14,15);
}

// B fragment 32x16 f16 for C = A * W^T with W[N,K] row-major:
// lane n%16 holds W[n, k+8*hf .. k+8*hf+7] and W[n, k+16+8*hf .. k+16+8*hf+7]
__device__ __forceinline__ v16h load_b_frag(const _Float16* __restrict__ wrow,
                                            int k, int hf) {
  v8h lo = *(const v8h*)(wrow + k + 8 * hf);
  v8h hi = *(const v8h*)(wrow + k + 16 + 8 * hf);
  return __builtin_shufflevector(lo, hi, 0,1,2,3,4,5,6,7,8,9,10,11,12,13,14,15);
}

// ---------------------------------------------------------------------------
// WMMA GEMM:  C[M,N] (fp32) = A[M,K] (f16, row-major) * W[N,K]^T (f16, row-major)
// block = 128 threads = 4 waves; block tile 128(M) x 64(N); wave tile 32 x 64.
// Per K-step: load 2 A frags + 4 B frags, then chain 8 WMMAs.
// grid = (M/128, N/64). K must be a multiple of 32.
// ---------------------------------------------------------------------------
__global__ void __launch_bounds__(128) gemm_wmma_k(const _Float16* __restrict__ A,
                                                   const _Float16* __restrict__ W,
                                                   float* __restrict__ C,
                                                   int K, int N) {
  const int wave  = threadIdx.x >> 5;
  const int lane  = threadIdx.x & 31;
  const int ln    = lane & 15;   // row/col-within-tile selector
  const int hf    = lane >> 4;   // half-wave selector
  const int mBase = blockIdx.x * 128 + wave * 32;
  const int nBase = blockIdx.y * 64;

  const _Float16* arow0 = A + (size_t)(mBase + ln) * K;
  const _Float16* arow1 = A + (size_t)(mBase + 16 + ln) * K;
  const _Float16* wrow0 = W + (size_t)(nBase + ln) * K;
  const _Float16* wrow1 = W + (size_t)(nBase + 16 + ln) * K;
  const _Float16* wrow2 = W + (size_t)(nBase + 32 + ln) * K;
  const _Float16* wrow3 = W + (size_t)(nBase + 48 + ln) * K;

  v8f acc[2][4] = {};

  for (int k = 0; k < K; k += 32) {
    if (k + 32 < K) {
      __builtin_prefetch(arow0 + k + 32, 0, 1);  // global_prefetch_b8
      __builtin_prefetch(wrow0 + k + 32, 0, 1);
    }
    // gather all fragments first so the loads form one clause and the
    // 8-WMMA chain below overlaps the next iteration's loads
    v16h a0 = load_a_frag(arow0, k, hf);
    v16h a1 = load_a_frag(arow1, k, hf);
    v16h b0 = load_b_frag(wrow0, k, hf);
    v16h b1 = load_b_frag(wrow1, k, hf);
    v16h b2 = load_b_frag(wrow2, k, hf);
    v16h b3 = load_b_frag(wrow3, k, hf);

    acc[0][0] = __builtin_amdgcn_wmma_f32_16x16x32_f16(false, a0, false, b0, (short)0, acc[0][0], false, false);
    acc[1][0] = __builtin_amdgcn_wmma_f32_16x16x32_f16(false, a1, false, b0, (short)0, acc[1][0], false, false);
    acc[0][1] = __builtin_amdgcn_wmma_f32_16x16x32_f16(false, a0, false, b1, (short)0, acc[0][1], false, false);
    acc[1][1] = __builtin_amdgcn_wmma_f32_16x16x32_f16(false, a1, false, b1, (short)0, acc[1][1], false, false);
    acc[0][2] = __builtin_amdgcn_wmma_f32_16x16x32_f16(false, a0, false, b2, (short)0, acc[0][2], false, false);
    acc[1][2] = __builtin_amdgcn_wmma_f32_16x16x32_f16(false, a1, false, b2, (short)0, acc[1][2], false, false);
    acc[0][3] = __builtin_amdgcn_wmma_f32_16x16x32_f16(false, a0, false, b3, (short)0, acc[0][3], false, false);
    acc[1][3] = __builtin_amdgcn_wmma_f32_16x16x32_f16(false, a1, false, b3, (short)0, acc[1][3], false, false);
  }

  // D layout: element j -> row tileM + 8*hf + j, col nBase + s*16 + ln
#pragma unroll
  for (int t = 0; t < 2; ++t) {
#pragma unroll
    for (int s = 0; s < 4; ++s) {
      float* cp = C + (size_t)(mBase + t * 16 + hf * 8) * N + (nBase + s * 16 + ln);
#pragma unroll
      for (int j = 0; j < 8; ++j) cp[(size_t)j * N] = acc[t][s][j];
    }
  }
}

// ---------------------------------------------------------------------------
// Elementwise / small kernels
// ---------------------------------------------------------------------------
__global__ void embed_k(const int* __restrict__ ids, const float* __restrict__ emb,
                        float* __restrict__ x) {
  int n = blockIdx.x, t = threadIdx.x;
  x[(size_t)n * DIMC + t] = emb[(size_t)ids[n] * DIMC + t] * 16.0f;  // sqrt(256)
}

__global__ void rmsnorm_k(const float* __restrict__ x, const float* __restrict__ w,
                          _Float16* __restrict__ out) {
  __shared__ float red[DIMC];
  int n = blockIdx.x, t = threadIdx.x;
  float v = x[(size_t)n * DIMC + t];
  red[t] = v * v;
  __syncthreads();
  for (int s = DIMC / 2; s > 0; s >>= 1) {
    if (t < s) red[t] += red[t + s];
    __syncthreads();
  }
  float sc = rsqrtf(red[0] / (float)DIMC + 1e-6f);
  out[(size_t)n * DIMC + t] = (_Float16)(v * sc * w[t]);
}

__global__ void cvt_k(const float* __restrict__ in, _Float16* __restrict__ out, int n) {
  int i = blockIdx.x * 256 + threadIdx.x;
  if (i < n) out[i] = (_Float16)in[i];
}

// causal-mask + row-normalize one layer's patterns: [H,S,S] -> [H,S,S]
__global__ void pnorm_k(const float* __restrict__ pat, float* __restrict__ pn) {
  __shared__ float red[SEQC];
  int hq = blockIdx.x;        // h*64 + q
  int q  = hq & (SEQC - 1);
  int t  = threadIdx.x;
  float p = (t <= q) ? pat[(size_t)hq * SEQC + t] : 0.0f;
  red[t] = p;
  __syncthreads();
  for (int s = SEQC / 2; s > 0; s >>= 1) {
    if (t < s) red[t] += red[t + s];
    __syncthreads();
  }
  pn[(size_t)hq * SEQC + t] = p / (red[0] + 1e-10f);
}

// compiled[b,q,d] = sum_k pnorm[h(d),q,k] * v[b,k,d]   (d = h*64+hd), f16 out
__global__ void compiled_k(const float* __restrict__ pn, const float* __restrict__ v,
                           _Float16* __restrict__ out) {
  int n = blockIdx.x;            // b*64+q
  int d = threadIdx.x;
  int h = d >> 6;
  int b = n >> 6, q = n & 63;
  const float* pr = pn + (size_t)h * SEQC * SEQC + (size_t)q * SEQC;
  const float* vb = v + (size_t)(b * SEQC) * DIMC + d;
  float acc = 0.0f;
#pragma unroll 4
  for (int k = 0; k < SEQC; ++k) acc += pr[k] * vb[(size_t)k * DIMC];
  out[(size_t)n * DIMC + d] = (_Float16)acc;
}

// single-head causal SDPA over HD=64; one block per (b,q), 64 threads
__global__ void refine_k(const float* __restrict__ rq, const float* __restrict__ rk,
                         const float* __restrict__ rv, _Float16* __restrict__ out) {
  __shared__ float qv[HD];
  __shared__ float p[SEQC];
  __shared__ float red[SEQC];
  int n = blockIdx.x;            // b*64+q
  int b = n >> 6, q = n & 63;
  int t = threadIdx.x;
  qv[t] = rq[(size_t)n * HD + t];
  __syncthreads();
  float s = -3.0e38f;
  if (t <= q) {
    const float* kr = rk + (size_t)(b * SEQC + t) * HD;
    float acc = 0.0f;
#pragma unroll 8
    for (int d = 0; d < HD; ++d) acc += qv[d] * kr[d];
    s = acc * 0.125f;            // 1/sqrt(64)
  }
  red[t] = s;
  __syncthreads();
  for (int w = SEQC / 2; w > 0; w >>= 1) {
    if (t < w) red[t] = fmaxf(red[t], red[t + w]);
    __syncthreads();
  }
  float m = red[0];
  __syncthreads();
  float e = (t <= q) ? expf(s - m) : 0.0f;
  p[t] = e;
  red[t] = e;
  __syncthreads();
  for (int w = SEQC / 2; w > 0; w >>= 1) {
    if (t < w) red[t] += red[t + w];
    __syncthreads();
  }
  float inv = 1.0f / red[0];
  float acc = 0.0f;
  for (int k = 0; k <= q; ++k) acc += p[k] * rv[(size_t)(b * SEQC + k) * HD + t];
  out[(size_t)n * HD + t] = (_Float16)(acc * inv);
}

__global__ void resid1_k(float* __restrict__ x, const float* __restrict__ a,
                         const float* __restrict__ r) {
  size_t i = (size_t)blockIdx.x * 256 + threadIdx.x;
  x[i] += a[i] + 0.5f * r[i];
}

__global__ void resid2_k(float* __restrict__ x, const float* __restrict__ a) {
  size_t i = (size_t)blockIdx.x * 256 + threadIdx.x;
  x[i] += a[i];
}

// gated = gelu_tanh(gate) * up, f16 out
__global__ void geglu_k(const float* __restrict__ g, const float* __restrict__ u,
                        _Float16* __restrict__ out) {
  size_t i = (size_t)blockIdx.x * 256 + threadIdx.x;
  float xg = g[i];
  float th = tanhf(0.7978845608f * (xg + 0.044715f * xg * xg * xg));
  out[i] = (_Float16)(0.5f * xg * (1.0f + th) * u[i]);
}

// ---------------------------------------------------------------------------
// Host side
// ---------------------------------------------------------------------------
static void launch_gemm(const _Float16* A, const _Float16* W, float* C,
                        int M, int N, int K, hipStream_t stream) {
  dim3 grid(M / 128, N / 64);
  gemm_wmma_k<<<grid, 128, 0, stream>>>(A, W, C, K, N);
}

static void launch_cvt(const float* in, _Float16* out, int n, hipStream_t stream) {
  cvt_k<<<(n + 255) / 256, 256, 0, stream>>>(in, out, n);
}

extern "C" void kernel_launch(void* const* d_in, const int* in_sizes, int n_in,
                              void* d_out, int out_size, void* d_ws, size_t ws_size,
                              hipStream_t stream) {
  const int*   input_ids  = (const int*)d_in[0];
  const float* embed      = (const float*)d_in[1];
  const float* patterns   = (const float*)d_in[2];
  const float* attn_nw    = (const float*)d_in[3];
  const float* v_w        = (const float*)d_in[4];
  const float* o_w        = (const float*)d_in[5];
  const float* rq_w       = (const float*)d_in[6];
  const float* rk_w       = (const float*)d_in[7];
  const float* rv_w       = (const float*)d_in[8];
  const float* ro_w       = (const float*)d_in[9];
  const float* ffn_nw     = (const float*)d_in[10];
  const float* gate_w     = (const float*)d_in[11];
  const float* up_w       = (const float*)d_in[12];
  const float* down_w     = (const float*)d_in[13];
  const float* norm_w     = (const float*)d_in[14];
  const float* lm_head_w  = (const float*)d_in[15];

  // ---- workspace carve-up (256B aligned slots) ----
  char* ws = (char*)d_ws;
  size_t off = 0;
  auto carve = [&](size_t bytes) -> char* {
    char* p = ws + off;
    off += (bytes + 255) & ~(size_t)255;
    return p;
  };
  float*    x      = (float*)   carve((size_t)TOK * DIMC * 4);
  _Float16* nh     = (_Float16*)carve((size_t)TOK * DIMC * 2);
  float*    vbuf   = (float*)   carve((size_t)TOK * DIMC * 4);
  float*    pnorm  = (float*)   carve((size_t)HEADS * SEQC * SEQC * 4);
  _Float16* comph  = (_Float16*)carve((size_t)TOK * DIMC * 2);
  float*    oout   = (float*)   carve((size_t)TOK * DIMC * 4);
  float*    rq     = (float*)   carve((size_t)TOK * HD * 4);
  float*    rk     = (float*)   carve((size_t)TOK * HD * 4);
  float*    rv     = (float*)   carve((size_t)TOK * HD * 4);
  _Float16* refh   = (_Float16*)carve((size_t)TOK * HD * 2);
  float*    roout  = (float*)   carve((size_t)TOK * DIMC * 4);
  float*    gatef  = (float*)   carve((size_t)TOK * FFC * 4);
  float*    upf    = (float*)   carve((size_t)TOK * FFC * 4);
  _Float16* gatedh = (_Float16*)carve((size_t)TOK * FFC * 2);
  float*    downf  = (float*)   carve((size_t)TOK * DIMC * 4);
  _Float16* xnh    = (_Float16*)carve((size_t)TOK * DIMC * 2);
  // per-layer f16 weight staging
  _Float16* wlv  = (_Float16*)carve((size_t)DIMC * DIMC * 2);
  _Float16* wlo  = (_Float16*)carve((size_t)DIMC * DIMC * 2);
  _Float16* wlq  = (_Float16*)carve((size_t)HD * DIMC * 2);
  _Float16* wlk  = (_Float16*)carve((size_t)HD * DIMC * 2);
  _Float16* wlvv = (_Float16*)carve((size_t)HD * DIMC * 2);
  _Float16* wlro = (_Float16*)carve((size_t)DIMC * HD * 2);
  _Float16* wlg  = (_Float16*)carve((size_t)FFC * DIMC * 2);
  _Float16* wlu  = (_Float16*)carve((size_t)FFC * DIMC * 2);
  _Float16* wld  = (_Float16*)carve((size_t)DIMC * FFC * 2);
  // lm_head f16 aliases the (no longer needed) gate fp32 buffer: 16MB <= 16.7MB
  _Float16* lmh = (_Float16*)gatef;

  // ---- embedding ----
  embed_k<<<TOK, DIMC, 0, stream>>>(input_ids, embed, x);

  // ---- layers ----
  for (int li = 0; li < NL; ++li) {
    // stage this layer's weights as f16
    launch_cvt(v_w  + (size_t)li * DIMC * DIMC, wlv,  DIMC * DIMC, stream);
    launch_cvt(o_w  + (size_t)li * DIMC * DIMC, wlo,  DIMC * DIMC, stream);
    launch_cvt(rq_w + (size_t)li * HD * DIMC,   wlq,  HD * DIMC,   stream);
    launch_cvt(rk_w + (size_t)li * HD * DIMC,   wlk,  HD * DIMC,   stream);
    launch_cvt(rv_w + (size_t)li * HD * DIMC,   wlvv, HD * DIMC,   stream);
    launch_cvt(ro_w + (size_t)li * DIMC * HD,   wlro, DIMC * HD,   stream);
    launch_cvt(gate_w + (size_t)li * FFC * DIMC, wlg, FFC * DIMC,  stream);
    launch_cvt(up_w   + (size_t)li * FFC * DIMC, wlu, FFC * DIMC,  stream);
    launch_cvt(down_w + (size_t)li * DIMC * FFC, wld, DIMC * FFC,  stream);

    // attn rmsnorm -> f16
    rmsnorm_k<<<TOK, DIMC, 0, stream>>>(x, attn_nw + (size_t)li * DIMC, nh);

    // v = normed @ v_w.T  (fp32 out for einsum)
    launch_gemm(nh, wlv, vbuf, TOK, DIMC, DIMC, stream);

    // pattern mixing (causal mask + renorm), then compiled = einsum(p, v)
    pnorm_k<<<HEADS * SEQC, SEQC, 0, stream>>>(
        patterns + (size_t)li * HEADS * SEQC * SEQC, pnorm);
    compiled_k<<<TOK, DIMC, 0, stream>>>(pnorm, vbuf, comph);

    // compiled @ o_w.T
    launch_gemm(comph, wlo, oout, TOK, DIMC, DIMC, stream);

    // refine head projections
    launch_gemm(nh, wlq,  rq, TOK, HD, DIMC, stream);
    launch_gemm(nh, wlk,  rk, TOK, HD, DIMC, stream);
    launch_gemm(nh, wlvv, rv, TOK, HD, DIMC, stream);

    // causal SDPA (single head)
    refine_k<<<TOK, SEQC, 0, stream>>>(rq, rk, rv, refh);

    // refine @ ro_w.T
    launch_gemm(refh, wlro, roout, TOK, DIMC, HD, stream);

    // x += compiled_o + 0.5*refine_o
    resid1_k<<<TOK, DIMC, 0, stream>>>(x, oout, roout);

    // FFN
    rmsnorm_k<<<TOK, DIMC, 0, stream>>>(x, ffn_nw + (size_t)li * DIMC, nh);
    launch_gemm(nh, wlg, gatef, TOK, FFC, DIMC, stream);
    launch_gemm(nh, wlu, upf,   TOK, FFC, DIMC, stream);
    geglu_k<<<(TOK * FFC) / 256, 256, 0, stream>>>(gatef, upf, gatedh);
    launch_gemm(gatedh, wld, downf, TOK, DIMC, FFC, stream);
    resid2_k<<<TOK, DIMC, 0, stream>>>(x, downf);
  }

  // ---- final norm + lm_head ----
  rmsnorm_k<<<TOK, DIMC, 0, stream>>>(x, norm_w, xnh);
  launch_cvt(lm_head_w, lmh, VOC * DIMC, stream);
  launch_gemm(xnh, lmh, (float*)d_out, TOK, VOC, DIMC, stream);

  (void)in_sizes; (void)n_in; (void)out_size; (void)ws_size;
}
